// Net1_64862596104438
// MI455X (gfx1250) — matrix operand order repaired and checked by reading, not compiled, not run on hardware
//
#include <hip/hip_runtime.h>

#define N_NODES 100000
#define F_DIM   128
#define N_EDGES 1600000

typedef __attribute__((ext_vector_type(16))) __bf16          v16bf;
typedef __attribute__((ext_vector_type(16))) unsigned short  v16u;
typedef __attribute__((ext_vector_type(8)))  float           v8f;

__device__ __forceinline__ unsigned short f2bf(float f) {
    unsigned int u = __builtin_bit_cast(unsigned int, f);
    u += 0x7FFFu + ((u >> 16) & 1u);          // round-to-nearest-even
    return (unsigned short)(u >> 16);
}

// ---------------- degree / normalization ----------------
__global__ void k_deg_init(float* __restrict__ deg) {
    int i = blockIdx.x * blockDim.x + threadIdx.x;
    if (i < N_NODES) deg[i] = 1.0f;           // self-loop
}

__global__ void k_deg_count(const int* __restrict__ dst, float* __restrict__ deg) {
    int e = blockIdx.x * blockDim.x + threadIdx.x;
    if (e < N_EDGES) atomicAdd(&deg[dst[e]], 1.0f);
}

__global__ void k_dinv(float* __restrict__ deg) {
    int i = blockIdx.x * blockDim.x + threadIdx.x;
    if (i < N_NODES) deg[i] = rsqrtf(deg[i]); // deg >= 1 always
}

// ---------------- dense transform: Y = X (Nx128) @ W (128x128), bf16 WMMA ----------------
__global__ void __launch_bounds__(256)
k_gemm128(const float* __restrict__ X, const float* __restrict__ W, float* __restrict__ Y) {
    const int lane = threadIdx.x & 31;
    const int wave = threadIdx.x >> 5;        // 8 waves -> 8 column tiles of 16
    const int row0 = blockIdx.x << 4;         // 16-row tile
    const int m    = lane & 15;
    const int half = lane >> 4;
    const int c0   = wave << 4;

    v8f acc = {};
#pragma unroll
    for (int kb = 0; kb < 128; kb += 32) {
        // A tile 16x32: lane holds row m; e<8 -> K=kb+half*8+e ; e>=8 -> K=kb+16+half*8+(e-8)
        const float* xr = X + (size_t)(row0 + m) * F_DIM + kb + half * 8;
        float4 a0 = *(const float4*)(xr);
        float4 a1 = *(const float4*)(xr + 4);
        float4 a2 = *(const float4*)(xr + 16);
        float4 a3 = *(const float4*)(xr + 20);
        v16u au;
        au[0]  = f2bf(a0.x); au[1]  = f2bf(a0.y); au[2]  = f2bf(a0.z); au[3]  = f2bf(a0.w);
        au[4]  = f2bf(a1.x); au[5]  = f2bf(a1.y); au[6]  = f2bf(a1.z); au[7]  = f2bf(a1.w);
        au[8]  = f2bf(a2.x); au[9]  = f2bf(a2.y); au[10] = f2bf(a2.z); au[11] = f2bf(a2.w);
        au[12] = f2bf(a3.x); au[13] = f2bf(a3.y); au[14] = f2bf(a3.z); au[15] = f2bf(a3.w);

        // B tile 32x16: lane holds column n=c0+m; e<8 -> K=kb+half*8+e ; e>=8 -> K=kb+16+half*8+(e-8)
        const float* wc = W + (size_t)(kb + half * 8) * F_DIM + (c0 + m);
        v16u bu;
#pragma unroll
        for (int e = 0; e < 16; ++e) bu[e] = f2bf(wc[(size_t)(e < 8 ? e : e + 8) * F_DIM]);

        v16bf av = __builtin_bit_cast(v16bf, au);
        v16bf bv = __builtin_bit_cast(v16bf, bu);
        acc = __builtin_amdgcn_wmma_f32_16x16x32_bf16(false, av, false, bv,
                                                      (short)0, acc, false, false);
    }
    // D layout: reg r, lane l -> (row0 + r + 8*half, c0 + m)
#pragma unroll
    for (int r = 0; r < 8; ++r)
        Y[(size_t)(row0 + r + 8 * half) * F_DIM + c0 + m] = acc[r];
}

// ---------------- self-loop init: agg = h * dinv^2 ----------------
__global__ void k_selfloop(const float* __restrict__ h, const float* __restrict__ dinv,
                           float* __restrict__ agg) {
    unsigned gid = blockIdx.x * blockDim.x + threadIdx.x;   // N*32 threads, float4 each
    unsigned node = gid >> 5, lane = gid & 31;
    if (node >= N_NODES) return;
    float dv  = dinv[node];
    float nrm = dv * dv;
    float4 v = *(const float4*)(h + (size_t)node * F_DIM + lane * 4);
    float4 o = make_float4(v.x * nrm, v.y * nrm, v.z * nrm, v.w * nrm);
    *(float4*)(agg + (size_t)node * F_DIM + lane * 4) = o;
}

// ---------------- edge scatter: agg[dst] += h[src] * dinv[src]*dinv[dst] ----------------
__global__ void k_edge(const int* __restrict__ src, const int* __restrict__ dst,
                       const float* __restrict__ dinv, const float* __restrict__ h,
                       float* __restrict__ agg) {
    unsigned gid = blockIdx.x * blockDim.x + threadIdx.x;   // E*32 threads
    unsigned e = gid >> 5, lane = gid & 31;
    if (e >= N_EDGES) return;
    int s = src[e], d = dst[e];
    float nrm = dinv[s] * dinv[d];
    float4 v = *(const float4*)(h + (size_t)s * F_DIM + lane * 4);
    float* o = agg + (size_t)d * F_DIM + lane * 4;
    atomicAdd(o + 0, v.x * nrm);
    atomicAdd(o + 1, v.y * nrm);
    atomicAdd(o + 2, v.z * nrm);
    atomicAdd(o + 3, v.w * nrm);
}

// ---------------- bias + tanh ----------------
__global__ void k_bias_tanh(const float* __restrict__ agg, const float* __restrict__ b,
                            float* __restrict__ o) {
    unsigned gid = blockIdx.x * blockDim.x + threadIdx.x;   // N*32 threads
    unsigned node = gid >> 5, lane = gid & 31;
    if (node >= N_NODES) return;
    float4 v  = *(const float4*)(agg + (size_t)node * F_DIM + lane * 4);
    float4 bb = *(const float4*)(b + lane * 4);
    float4 r  = make_float4(tanhf(v.x + bb.x), tanhf(v.y + bb.y),
                            tanhf(v.z + bb.z), tanhf(v.w + bb.w));
    *(float4*)(o + (size_t)node * F_DIM + lane * 4) = r;
}

// ---------------- readout: out = sigmoid(emb @ Wfc + bfc), wave per node ----------------
__global__ void k_fc_sigmoid(const float* __restrict__ emb, const float* __restrict__ Wfc,
                             const float* __restrict__ bfc, float* __restrict__ out) {
    unsigned gid = blockIdx.x * blockDim.x + threadIdx.x;   // N*32 threads
    unsigned node = gid >> 5, lane = gid & 31;
    if (node >= N_NODES) return;
    float4 v = *(const float4*)(emb + (size_t)node * F_DIM + lane * 4);
    float4 w = *(const float4*)(Wfc + lane * 4);
    float s = v.x * w.x + v.y * w.y + v.z * w.z + v.w * w.w;
#pragma unroll
    for (int off = 16; off > 0; off >>= 1) s += __shfl_xor(s, off, 32);
    if (lane == 0) out[node] = 1.0f / (1.0f + expf(-(s + bfc[0])));
}

extern "C" void kernel_launch(void* const* d_in, const int* in_sizes, int n_in,
                              void* d_out, int out_size, void* d_ws, size_t ws_size,
                              hipStream_t stream) {
    const float* x   = (const float*)d_in[0];
    const int*   ei  = (const int*)  d_in[1];   // [2, E] int32
    const float* W1  = (const float*)d_in[2];
    const float* b1  = (const float*)d_in[3];
    const float* W2  = (const float*)d_in[4];
    const float* b2  = (const float*)d_in[5];
    const float* Wfc = (const float*)d_in[6];
    const float* bfc = (const float*)d_in[7];
    const int* src = ei;
    const int* dst = ei + N_EDGES;

    float* out = (float*)d_out;                 // [N] first output
    float* emb = out + N_NODES;                 // [N,128] second output

    char*  ws   = (char*)d_ws;
    float* dinv = (float*)ws;                                              // N floats
    size_t off  = (((size_t)N_NODES * 4) + 255) & ~(size_t)255;
    float* bufA = (float*)(ws + off);                                      // N*128
    float* bufB = bufA + (size_t)N_NODES * F_DIM;                          // N*128

    const int TB = 256;
    const int gN   = (N_NODES + TB - 1) / TB;
    const int gE   = (N_EDGES + TB - 1) / TB;
    const int gN32 = (N_NODES * 32 + TB - 1) / TB;
    const unsigned gE32 = (unsigned)(((size_t)N_EDGES * 32 + TB - 1) / TB);
    const int gT   = N_NODES / 16;              // 100000 / 16 = 6250 exact

    // normalization
    k_deg_init <<<gN, TB, 0, stream>>>(dinv);
    k_deg_count<<<gE, TB, 0, stream>>>(dst, dinv);
    k_dinv     <<<gN, TB, 0, stream>>>(dinv);

    // layer 1: h1 = X@W1 ; agg = D^-1/2 A_hat D^-1/2 h1 ; hA = tanh(agg + b1)
    k_gemm128  <<<gT,  256, 0, stream>>>(x, W1, bufA);
    k_selfloop <<<gN32, TB, 0, stream>>>(bufA, dinv, bufB);
    k_edge     <<<gE32, TB, 0, stream>>>(src, dst, dinv, bufA, bufB);
    k_bias_tanh<<<gN32, TB, 0, stream>>>(bufB, b1, bufA);

    // layer 2: h2 = hA@W2 ; agg ; emb = tanh(agg + b2)
    k_gemm128  <<<gT,  256, 0, stream>>>(bufA, W2, bufB);
    k_selfloop <<<gN32, TB, 0, stream>>>(bufB, dinv, bufA);
    k_edge     <<<gE32, TB, 0, stream>>>(src, dst, dinv, bufB, bufA);
    k_bias_tanh<<<gN32, TB, 0, stream>>>(bufA, b2, emb);

    // readout
    k_fc_sigmoid<<<gN32, TB, 0, stream>>>(emb, Wfc, bfc, out);
}
